// RecurrentLayer_29618094473421
// MI455X (gfx1250) — matrix operand order repaired and checked by reading, not compiled, not run on hardware
//
#include <hip/hip_runtime.h>
#include <hip/hip_bf16.h>

#define HIDN 512
#define INDIM 256
#define BATCH 64
#define TLEN 1024

typedef __attribute__((ext_vector_type(16))) _Float16 v16h;
typedef __attribute__((ext_vector_type(8)))  _Float16 v8h;
typedef __attribute__((ext_vector_type(8)))  float    v8f;
typedef __attribute__((ext_vector_type(4)))  float    v4f;
typedef __attribute__((ext_vector_type(4)))  unsigned int u32x4;
typedef __attribute__((ext_vector_type(4)))  int      i32x4;
typedef __attribute__((ext_vector_type(8)))  int      i32x8;

#if defined(__has_builtin)
#  if __has_builtin(__builtin_amdgcn_tensor_load_to_lds) && \
      __has_builtin(__builtin_amdgcn_s_wait_tensorcnt)
#    define HAVE_TDM 1
#  endif
#endif
#ifndef HAVE_TDM
#  define HAVE_TDM 0
#endif

// ---------------------------------------------------------------------------
// 16x32 f16 A-fragment, contiguous k order used here (wave32):
//   lane = lh*16 + m  (m = row 0..15, lh = half-wave 0/1)
//   element e    -> K = kBase + lh*16 + e
// ---------------------------------------------------------------------------
static __device__ inline v16h load_a_frag_f32(const float* __restrict__ rowPtr,
                                              int kBase, int lh) {
  const v4f f0 = *(const v4f*)(rowPtr + kBase + lh * 16);
  const v4f f1 = *(const v4f*)(rowPtr + kBase + lh * 16 + 4);
  const v4f f2 = *(const v4f*)(rowPtr + kBase + lh * 16 + 8);
  const v4f f3 = *(const v4f*)(rowPtr + kBase + lh * 16 + 12);
  v16h a;
#pragma unroll
  for (int e = 0; e < 4; ++e) {
    a[e]      = (_Float16)f0[e];
    a[e + 4]  = (_Float16)f1[e];
    a[e + 8]  = (_Float16)f2[e];
    a[e + 12] = (_Float16)f3[e];
  }
  return a;
}

static __device__ inline v16h load_a_frag_f16(const _Float16* rowPtr, int kBase,
                                              int lh) {
  const v8h lo = *(const v8h*)(rowPtr + kBase + lh * 16);
  const v8h hi = *(const v8h*)(rowPtr + kBase + lh * 16 + 8);
  v16h a;
#pragma unroll
  for (int e = 0; e < 8; ++e) {
    a[e]     = lo[e];
    a[e + 8] = hi[e];
  }
  return a;
}

#if HAVE_TDM
// ---------------------------------------------------------------------------
// TDM descriptor (ISA 08_async_tensor §8): 2D fp32 tile, 16 rows x 512 cols,
// row stride T*HID elements, packed into LDS at ldsOff.
// ---------------------------------------------------------------------------
static __device__ inline void tdm_load_vin(const float* __restrict__ v_in,
                                           int bBase, int t,
                                           unsigned int ldsOff) {
  const unsigned long long gaddr =
      (unsigned long long)(v_in + ((size_t)bBase * TLEN + t) * HIDN);
  u32x4 g0;
  g0[0] = 1u;                                         // count=1, user descriptor
  g0[1] = ldsOff;                                     // lds_addr (bytes)
  g0[2] = (unsigned int)gaddr;                        // global_addr[31:0]
  g0[3] = (unsigned int)(gaddr >> 32) | (2u << 30);   // addr[56:32] | type=2
  i32x8 g1;
  g1[0] = 0x20000;                                    // data_size = 4 bytes
  g1[1] = (int)(((unsigned)HIDN) << 16);              // tensor_dim0 = 512
  g1[2] = (int)(16u << 16);                           // tensor_dim1 = 16
  g1[3] = (int)(((unsigned)HIDN) << 16);              // tile_dim0  = 512
  g1[4] = 16;                                         // tile_dim1  = 16
  g1[5] = TLEN * HIDN;                                // tensor_dim0_stride
  g1[6] = 0;
  g1[7] = 0;
  const i32x4 z4 = {0, 0, 0, 0};
#if __clang_major__ >= 23
  const i32x8 z8 = {0, 0, 0, 0, 0, 0, 0, 0};
  __builtin_amdgcn_tensor_load_to_lds(g0, g1, z4, z4, z8, 0);
#else
  __builtin_amdgcn_tensor_load_to_lds(g0, g1, z4, z4, 0);
#endif
}
#endif

// ---------------------------------------------------------------------------
// Kernel 1: one-time weight transpose + f32->f16 convert.
// ---------------------------------------------------------------------------
__global__ __launch_bounds__(256) void prep_weights_kernel(
    const float* __restrict__ W_in, const float* __restrict__ W_hid,
    _Float16* __restrict__ winT, _Float16* __restrict__ whT) {
  const int idx = blockIdx.x * 256 + threadIdx.x;
  if (idx < HIDN * HIDN) {
    const int n = idx & (HIDN - 1);
    const int k = idx >> 9;
    whT[(size_t)n * HIDN + k] = (_Float16)W_hid[(size_t)k * HIDN + n];
  } else {
    const int j = idx - HIDN * HIDN;
    const int n = j & (HIDN - 1);
    const int k = j >> 9;  // 0..255
    winT[(size_t)n * INDIM + k] = (_Float16)W_in[(size_t)k * HIDN + n];
  }
}

// ---------------------------------------------------------------------------
// Kernel 2: input projection  v_in = x @ W_in + b_in   (M=65536,N=512,K=256)
// Block = 256 threads (8 waves); block tile 128(M)x64(N); wave = 4 WMMA tiles.
// ---------------------------------------------------------------------------
__global__ __launch_bounds__(256) void proj_kernel(
    const float* __restrict__ x, const float* __restrict__ b_in,
    const _Float16* __restrict__ winT, float* __restrict__ v_in) {
  const int tid  = threadIdx.x;
  const int lane = tid & 31;
  const int w    = tid >> 5;
  const int lh   = lane >> 4;
  const int ln   = lane & 15;
  const int mBase = blockIdx.x * 128 + w * 16;
  const int nBase = blockIdx.y * 64;

  const float* aRow = x + (size_t)(mBase + ln) * INDIM;

  v8f c[4] = {};
  for (int kk = 0; kk < INDIM; kk += 32) {
    const v16h a = load_a_frag_f32(aRow, kk, lh);
#pragma unroll
    for (int i = 0; i < 4; ++i) {
      const int n = nBase + i * 16 + ln;
      const v16h b = *(const v16h*)(winT + (size_t)n * INDIM + kk + lh * 16);
      c[i] = __builtin_amdgcn_wmma_f32_16x16x32_f16(false, a, false, b,
                                                    (short)0, c[i], false, false);
    }
  }

#pragma unroll
  for (int i = 0; i < 4; ++i) {
    const int n = nBase + i * 16 + ln;
    const float bias = b_in[n];
#pragma unroll
    for (int r = 0; r < 8; ++r) {
      const int m = mBase + r + lh * 8;  // C/D layout: VGPR r -> M=r (+8 upper half)
      v_in[(size_t)m * HIDN + n] = c[i][r] + bias;
    }
  }
}

// ---------------------------------------------------------------------------
// Kernel 3: persistent sequential scan.  Grid = B/16 = 4 blocks, 256 threads.
// - v-state lives in REGISTERS (fixed owner lane per element).
// - fr (f16) double-buffered in LDS (A operand of the recurrent GEMM).
// - v_in forcing term prefetched one full step ahead by wave 0 via the
//   Tensor Data Mover into a double-buffered LDS tile (TENSORcnt-tracked).
// - Recurrent GEMM software-pipelined: next K-step's B fragments are in
//   flight while current WMMAs execute.
// Dynamic LDS: fr 2x16KB + vin 2x32KB = 96 KB (<< 320 KB/WGP).
// ---------------------------------------------------------------------------
__global__ __launch_bounds__(256) void scan_kernel(
    const float* __restrict__ init_state, const float* __restrict__ b_hid,
    const float* __restrict__ alpha, const _Float16* __restrict__ whT,
    const float* __restrict__ v_in, float* __restrict__ out) {
  extern __shared__ __align__(16) char smem[];
  // Individual scalar pointers (NOT pointer arrays: aggregate initializers
  // holding LDS addrspacecasts become unsupported static initializers).
  _Float16* const fr0  = (_Float16*)smem;
  _Float16* const fr1  = (_Float16*)(smem + 16 * 1024);
  float*    const vin0 = (float*)(smem + 32 * 1024);
  float*    const vin1 = (float*)(smem + 64 * 1024);

  const int tid   = threadIdx.x;
  const int lane  = tid & 31;
  const int w     = tid >> 5;
  const int lh    = lane >> 4;
  const int ln    = lane & 15;
  const int bBase = blockIdx.x * 16;

  int nIdx[4];
  float al[4], bh[4];
#pragma unroll
  for (int i = 0; i < 4; ++i) {
    nIdx[i] = w * 64 + i * 16 + ln;
    al[i] = alpha[nIdx[i]];
    bh[i] = b_hid[nIdx[i]];
  }

  // fr0 = relu(v0) quantized to f16 into buffer 0 (cooperative, coalesced).
#pragma unroll 4
  for (int j = 0; j < 32; ++j) {
    const int idx = tid * 32 + j;
    const int m = idx >> 9;
    const int n = idx & (HIDN - 1);
    fr0[idx] = (_Float16)fmaxf(init_state[(size_t)(bBase + m) * HIDN + n], 0.0f);
  }

  // v-state in registers: element (m = r + lh*8, n = nIdx[i]) owned by this lane.
  float vreg[4][8];
#pragma unroll
  for (int i = 0; i < 4; ++i)
#pragma unroll
    for (int r = 0; r < 8; ++r)
      vreg[i][r] = init_state[(size_t)(bBase + r + lh * 8) * HIDN + nIdx[i]];

#if HAVE_TDM
  if (w == 0) {
    tdm_load_vin(v_in, bBase, 0, (unsigned int)(unsigned long long)vin0);
    __builtin_amdgcn_s_wait_tensorcnt(0);
  }
#endif
  __syncthreads();

  for (int t = 0; t < TLEN; ++t) {
    const _Float16* frCur = (t & 1) ? fr1 : fr0;
    _Float16*       frNxt = (t & 1) ? fr0 : fr1;
    const float*    vinCur = (t & 1) ? vin1 : vin0;
    float*          vinNxt = (t & 1) ? vin0 : vin1;

#if HAVE_TDM
    // Kick off next step's forcing-term DMA; completes under this step's GEMM.
    if (w == 0 && t + 1 < TLEN)
      tdm_load_vin(v_in, bBase, t + 1, (unsigned int)(unsigned long long)vinNxt);
#else
    (void)vinCur; (void)vinNxt;
    float vin[4][8];
#pragma unroll
    for (int i = 0; i < 4; ++i)
#pragma unroll
      for (int r = 0; r < 8; ++r)
        vin[i][r] =
            v_in[((size_t)(bBase + r + lh * 8) * TLEN + t) * HIDN + nIdx[i]];
#endif

    // v_hid = fr @ W_hid : K=512, software-pipelined B fragments.
    v8f c[4] = {};
    const _Float16* aRow = frCur + ln * HIDN;
    v16h bcur[4];
#pragma unroll
    for (int i = 0; i < 4; ++i)
      bcur[i] = *(const v16h*)(whT + (size_t)nIdx[i] * HIDN + lh * 16);
#pragma unroll
    for (int kk = 0; kk < HIDN; kk += 32) {
      const v16h a = load_a_frag_f16(aRow, kk, lh);
      v16h bnext[4];
      if (kk + 32 < HIDN) {
#pragma unroll
        for (int i = 0; i < 4; ++i)
          bnext[i] =
              *(const v16h*)(whT + (size_t)nIdx[i] * HIDN + kk + 32 + lh * 16);
      }
#pragma unroll
      for (int i = 0; i < 4; ++i)
        c[i] = __builtin_amdgcn_wmma_f32_16x16x32_f16(false, a, false, bcur[i],
                                                      (short)0, c[i], false,
                                                      false);
      if (kk + 32 < HIDN) {
#pragma unroll
        for (int i = 0; i < 4; ++i) bcur[i] = bnext[i];
      }
    }

    // Fused epilogue: leaky integrate + ReLU; state stays in registers.
#pragma unroll
    for (int i = 0; i < 4; ++i) {
      const int n = nIdx[i];
#pragma unroll
      for (int r = 0; r < 8; ++r) {
        const int m = r + lh * 8;
#if HAVE_TDM
        const float vi = vinCur[m * HIDN + n];
#else
        const float vi = vin[i][r];
#endif
        const float s  = c[i][r] + bh[i] + vi;          // v_hid + b_hid + v_in
        const float vn = vreg[i][r] + al[i] * (s - vreg[i][r]);
        const float fr = fmaxf(vn, 0.0f);
        vreg[i][r] = vn;
        out[((size_t)(bBase + m) * TLEN + t) * HIDN + n] = fr;
        frNxt[m * HIDN + n] = (_Float16)fr;
      }
#if !HAVE_TDM
      if (t + 1 < TLEN)
        __builtin_prefetch(
            &v_in[((size_t)(bBase + lh * 8) * TLEN + (t + 1)) * HIDN + n], 0, 1);
#endif
    }

#if HAVE_TDM
    if (w == 0) __builtin_amdgcn_s_wait_tensorcnt(0);
#endif
    __syncthreads();  // fr[t+1] (and vin[t+1] DMA) visible before next step
  }
}

// ---------------------------------------------------------------------------
// Workspace layout:
//   [0,       512KB)  W_hidT  f16 (512x512)
//   [512KB,   768KB)  W_inT   f16 (512x256)
//   [768KB, ~129MB)   v_in    f32 (B*T*HID)
// ---------------------------------------------------------------------------
extern "C" void kernel_launch(void* const* d_in, const int* in_sizes, int n_in,
                              void* d_out, int out_size, void* d_ws, size_t ws_size,
                              hipStream_t stream) {
  (void)in_sizes; (void)n_in; (void)out_size; (void)ws_size;
  const float* x          = (const float*)d_in[0];
  const float* init_state = (const float*)d_in[1];
  const float* W_in       = (const float*)d_in[2];
  const float* b_in       = (const float*)d_in[3];
  const float* W_hid      = (const float*)d_in[4];
  const float* b_hid      = (const float*)d_in[5];
  const float* alpha      = (const float*)d_in[6];
  float* out = (float*)d_out;

  char* ws = (char*)d_ws;
  _Float16* whT  = (_Float16*)ws;                 // 512 KB
  _Float16* winT = (_Float16*)(ws + 512 * 1024);  // 256 KB
  float*    v_in = (float*)(ws + 768 * 1024);     // 128 MB

  prep_weights_kernel<<<1536, 256, 0, stream>>>(W_in, W_hid, winT, whT);

  dim3 pgrid(512, 8);
  proj_kernel<<<pgrid, 256, 0, stream>>>(x, b_in, winT, v_in);

  scan_kernel<<<BATCH / 16, 256, 96 * 1024, stream>>>(init_state, b_hid, alpha,
                                                      whT, v_in, out);
}
